// QuantizedMultiHeadAttention_7043746365915
// MI455X (gfx1250) — matrix operand order repaired and checked by reading, not compiled, not run on hardware
//
#include <hip/hip_runtime.h>
#include <hip/hip_bf16.h>
#include <math.h>

// ---------------------------------------------------------------------------
// Fused QuantizedMultiHeadAttention for gfx1250 (MI455X), wave32 WMMA path.
// Requires ws_size >= ~64.1 MB.
// ---------------------------------------------------------------------------

typedef __attribute__((ext_vector_type(16))) _Float16 v16h;
typedef __attribute__((ext_vector_type(8)))  float    v8f;

constexpr int B_  = 2;
constexpr int S_  = 2048;
constexpr int E_  = 1024;
constexpr int H_  = 16;
constexpr int D_  = 64;
constexpr int M_  = B_ * S_;            // 4096 rows in projection GEMMs
constexpr float SCALE_ = 0.125f;        // 1/sqrt(64)
constexpr int WELEM = E_ * E_;          // 1048576 per weight matrix
constexpr int XELEM = M_ * E_;          // 4194304 per activation tensor

// ----------------------------- scale / quant -------------------------------

__global__ void init_scales_kernel(unsigned int* sc) {
    if (threadIdx.x < 4) sc[threadIdx.x] = 0u;
}

__global__ void absmax_kernel(const float* __restrict__ w0, const float* __restrict__ w1,
                              const float* __restrict__ w2, const float* __restrict__ w3,
                              unsigned int* __restrict__ sc, int n) {
    const float* w = (blockIdx.y == 0) ? w0 : (blockIdx.y == 1) ? w1
                   : (blockIdx.y == 2) ? w2 : w3;
    __shared__ float red[256];
    float m = 0.0f;
    for (int i = blockIdx.x * blockDim.x + threadIdx.x; i < n; i += gridDim.x * blockDim.x)
        m = fmaxf(m, fabsf(w[i]));
    red[threadIdx.x] = m;
    __syncthreads();
    for (int s = 128; s > 0; s >>= 1) {
        if ((int)threadIdx.x < s) red[threadIdx.x] = fmaxf(red[threadIdx.x], red[threadIdx.x + s]);
        __syncthreads();
    }
    if (threadIdx.x == 0)
        atomicMax(&sc[blockIdx.y], __float_as_uint(red[0]));  // floats >= 0: bit order == value order
}

// quantize -> dequantize weights (per-tensor symmetric int8), emit f16
__global__ void dequant_kernel(const float* __restrict__ w0, const float* __restrict__ w1,
                               const float* __restrict__ w2, const float* __restrict__ w3,
                               const unsigned int* __restrict__ sc,
                               _Float16* __restrict__ out, int n) {
    const int t = blockIdx.y;
    const float* w = (t == 0) ? w0 : (t == 1) ? w1 : (t == 2) ? w2 : w3;
    const float amax = __uint_as_float(sc[t]);
    const float s    = amax * (1.0f / 127.0f);
    const float inv  = (s > 0.0f) ? (1.0f / s) : 0.0f;
    const int i = blockIdx.x * blockDim.x + threadIdx.x;
    if (i < n) {
        float q = rintf(w[i] * inv);               // round half to even, like jnp.round
        q = fminf(fmaxf(q, -128.0f), 127.0f);
        out[(size_t)t * n + i] = (_Float16)(q * s);
    }
}

__global__ void f2h_kernel(const float* __restrict__ a, const float* __restrict__ b,
                           const float* __restrict__ c, _Float16* __restrict__ out, int n) {
    const int t = blockIdx.y;
    const float* src = (t == 0) ? a : (t == 1) ? b : c;
    const int i = blockIdx.x * blockDim.x + threadIdx.x;
    if (i < n) out[(size_t)t * n + i] = (_Float16)src[i];
}

// ------------------------- DPP 16-lane row reductions -----------------------
// DPP rows are 16 lanes on this ISA == exactly one C-fragment lane group.
template <int CTRL>
__device__ __forceinline__ float dpp_xmov(float x) {
    return __int_as_float(__builtin_amdgcn_update_dpp(
        0, __float_as_int(x), CTRL, 0xF, 0xF, true));
}
__device__ __forceinline__ float row16_max(float v) {
    v = fmaxf(v, dpp_xmov<0xB1>(v));    // quad_perm [1,0,3,2]
    v = fmaxf(v, dpp_xmov<0x4E>(v));    // quad_perm [2,3,0,1]
    v = fmaxf(v, dpp_xmov<0x141>(v));   // row_half_mirror
    v = fmaxf(v, dpp_xmov<0x140>(v));   // row_mirror
    return v;
}
__device__ __forceinline__ float row16_sum(float v) {
    v += dpp_xmov<0xB1>(v);
    v += dpp_xmov<0x4E>(v);
    v += dpp_xmov<0x141>(v);
    v += dpp_xmov<0x140>(v);
    return v;
}

// --------------------------- WMMA fragment helpers -------------------------
// A-matrix 16x32 f16 (ISA 7.12.2): lane<16 row=lane, halves0-7=K0..7, 8-15=K16..23;
// lane>=16 row=lane-16, halves0-7=K8..15, 8-15=K24..31.
__device__ __forceinline__ v16h load_a_frag(const _Float16* __restrict__ base, int ld) {
    const int lane = threadIdx.x & 31;
    const _Float16* r = base + (size_t)(lane & 15) * ld + ((lane >> 4) ? 8 : 0);
    v16h a;
#pragma unroll
    for (int i = 0; i < 8; ++i) { a[i] = r[i]; a[8 + i] = r[16 + i]; }
    return a;
}

// B-matrix 32x16 f16 fed from a row-major [N][K] buffer (column n of B = row n):
// lane<16: n=lane, K=0..7,16..23; lane>=16: n=lane-16, K=8..15,24..31 (matches A-frag).
__device__ __forceinline__ v16h load_bT_frag(const _Float16* __restrict__ base, int ld) {
    const int lane = threadIdx.x & 31;
    const _Float16* r = base + (size_t)(lane & 15) * ld + ((lane >> 4) ? 8 : 0);
    v16h b;
#pragma unroll
    for (int i = 0; i < 16; ++i) b[i] = r[i < 8 ? i : i + 8];
    return b;
}

// ------------------------------- GEMM kernel -------------------------------
// Y[m][n] = sum_k A[m][k] * Wt[n][k] + bias[n]
// One wave computes a 32x64 output tile: 2 A-frags x 4 B-frags -> 8 WMMAs/K-step.
// MODE 0: store f16, head-major [B,H,S,D] (n -> (h,d))
// MODE 1: store f32, row-major [M][N] (final output)
template <int MODE>
__global__ void wmma_gemm_kernel(const _Float16* __restrict__ A,
                                 const _Float16* __restrict__ Wt,
                                 const float* __restrict__ bias,
                                 void* __restrict__ outp, int K, int N) {
    const int lane = threadIdx.x & 31;
    const int grp  = lane >> 4;
    const int nl   = lane & 15;
    const int n0   = blockIdx.x * 64;
    const int m0   = blockIdx.y * 32;

    const _Float16* arow0 = A + (size_t)(m0 + nl) * K + (grp ? 8 : 0);
    const _Float16* arow1 = arow0 + (size_t)16 * K;
    const _Float16* brow[4];
#pragma unroll
    for (int ni = 0; ni < 4; ++ni)
        brow[ni] = Wt + (size_t)(n0 + ni * 16 + nl) * K + (grp ? 8 : 0);

    v8f c[2][4];
#pragma unroll
    for (int mi = 0; mi < 2; ++mi)
#pragma unroll
        for (int ni = 0; ni < 4; ++ni) c[mi][ni] = v8f{};

    for (int k0 = 0; k0 < K; k0 += 32) {
        __builtin_prefetch((const void*)(arow0 + k0 + 64), 0, 1);
        __builtin_prefetch((const void*)(brow[0] + k0 + 64), 0, 1);
        v16h a0, a1;
#pragma unroll
        for (int i = 0; i < 8; ++i) {
            a0[i] = arow0[k0 + i]; a0[8 + i] = arow0[k0 + 16 + i];
            a1[i] = arow1[k0 + i]; a1[8 + i] = arow1[k0 + 16 + i];
        }
#pragma unroll
        for (int ni = 0; ni < 4; ++ni) {
            v16h b;
#pragma unroll
            for (int i = 0; i < 16; ++i) b[i] = brow[ni][k0 + (i < 8 ? i : i + 8)];
            c[0][ni] = __builtin_amdgcn_wmma_f32_16x16x32_f16(false, a0, false, b, (short)0, c[0][ni], false, false);
            c[1][ni] = __builtin_amdgcn_wmma_f32_16x16x32_f16(false, a1, false, b, (short)0, c[1][ni], false, false);
        }
    }

#pragma unroll
    for (int ni = 0; ni < 4; ++ni) {
        const int col = n0 + ni * 16 + nl;
        const float bv = bias[col];
        if (MODE == 0) {
            _Float16* o = (_Float16*)outp;
            const int h = col >> 6, d = col & 63;
#pragma unroll
            for (int mi = 0; mi < 2; ++mi)
#pragma unroll
                for (int e = 0; e < 8; ++e) {
                    const int row = m0 + mi * 16 + e + 8 * grp;    // global row in [B*S]
                    const int bb = row >> 11, s = row & (S_ - 1);
                    o[(((size_t)bb * H_ + h) * S_ + s) * D_ + d] = (_Float16)(c[mi][ni][e] + bv);
                }
        } else {
            float* o = (float*)outp;
#pragma unroll
            for (int mi = 0; mi < 2; ++mi)
#pragma unroll
                for (int e = 0; e < 8; ++e) {
                    const int row = m0 + mi * 16 + e + 8 * grp;
                    o[(size_t)row * N + col] = c[mi][ni][e] + bv;
                }
        }
    }
}

// --------------------------- flash attention core --------------------------
// One wave per (b, h, 16-row q tile). Online softmax over kv chunks of 32.
__global__ void flash_attn_kernel(const _Float16* __restrict__ Qh,
                                  const _Float16* __restrict__ Kh,
                                  const _Float16* __restrict__ Vh,
                                  _Float16* __restrict__ Oh /* [B,S,E] f16 */) {
    const int lane = threadIdx.x & 31;
    const int grp  = lane >> 4;
    const int nl   = lane & 15;
    const int q0   = blockIdx.x * 16;
    const int h    = blockIdx.y;
    const int b    = blockIdx.z;
    const size_t base = ((size_t)b * H_ + h) * S_ * D_;
    const _Float16* Q  = Qh + base;
    const _Float16* Kp = Kh + base;
    const _Float16* Vp = Vh + base;

    // Q tile as two A-frags covering d = 0..31 and 32..63
    const v16h qa0 = load_a_frag(Q + (size_t)q0 * D_,      D_);
    const v16h qa1 = load_a_frag(Q + (size_t)q0 * D_ + 32, D_);

    v8f acc[4] = {v8f{}, v8f{}, v8f{}, v8f{}};
    float mrow[8], lrow[8];
#pragma unroll
    for (int e = 0; e < 8; ++e) { mrow[e] = -1e30f; lrow[e] = 0.0f; }

    __shared__ _Float16 ptile[16 * 32];

    for (int kv = 0; kv < S_; kv += 32) {
        // ---- scores = Q K^T for 32 keys (two 16-wide column groups) ----
        v8f s0 = {}, s1 = {};
        {
            const v16h b00 = load_bT_frag(Kp + (size_t)kv * D_,             D_); // keys kv..+15, d 0..31
            const v16h b01 = load_bT_frag(Kp + (size_t)kv * D_ + 32,        D_); // keys kv..+15, d 32..63
            const v16h b10 = load_bT_frag(Kp + (size_t)(kv + 16) * D_,      D_);
            const v16h b11 = load_bT_frag(Kp + (size_t)(kv + 16) * D_ + 32, D_);
            s0 = __builtin_amdgcn_wmma_f32_16x16x32_f16(false, qa0, false, b00, (short)0, s0, false, false);
            s0 = __builtin_amdgcn_wmma_f32_16x16x32_f16(false, qa1, false, b01, (short)0, s0, false, false);
            s1 = __builtin_amdgcn_wmma_f32_16x16x32_f16(false, qa0, false, b10, (short)0, s1, false, false);
            s1 = __builtin_amdgcn_wmma_f32_16x16x32_f16(false, qa1, false, b11, (short)0, s1, false, false);
        }

        // ---- online softmax: row stats via DPP 16-lane reductions ----
#pragma unroll
        for (int e = 0; e < 8; ++e) {
            const float v0 = s0[e] * SCALE_;
            const float v1 = s1[e] * SCALE_;
            const float mx = row16_max(fmaxf(v0, v1));
            const float mnew = fmaxf(mrow[e], mx);
            const float p0 = __expf(v0 - mnew);
            const float p1 = __expf(v1 - mnew);
            const float rs = row16_sum(p0 + p1);
            const float corr = __expf(mrow[e] - mnew);
            lrow[e] = lrow[e] * corr + rs;
            mrow[e] = mnew;
#pragma unroll
            for (int t = 0; t < 4; ++t) acc[t][e] *= corr;
            const int row = e + 8 * grp;               // C-layout row owned by this lane group
            ptile[row * 32 + nl]      = (_Float16)p0;
            ptile[row * 32 + 16 + nl] = (_Float16)p1;
        }
        __syncthreads();

        // ---- P (16x32) back as an A-frag from LDS ----
        const _Float16* pr = &ptile[nl * 32 + (grp ? 8 : 0)];
        v16h pa;
#pragma unroll
        for (int i = 0; i < 8; ++i) { pa[i] = pr[i]; pa[8 + i] = pr[16 + i]; }
        __syncthreads();

        // ---- acc += P * V (V tile 32x64, four 16-col B frags, strided cols) ----
#pragma unroll
        for (int t = 0; t < 4; ++t) {
            const _Float16* vp = Vp + (size_t)(kv + (grp ? 8 : 0)) * D_ + t * 16 + nl;
            v16h vb;
#pragma unroll
            for (int i = 0; i < 16; ++i) vb[i] = vp[(size_t)(i < 8 ? i : i + 8) * D_];
            acc[t] = __builtin_amdgcn_wmma_f32_16x16x32_f16(false, pa, false, vb, (short)0, acc[t], false, false);
        }
    }

    // ---- normalize and scatter back to [B,S,E] f16 ----
#pragma unroll
    for (int t = 0; t < 4; ++t) {
#pragma unroll
        for (int e = 0; e < 8; ++e) {
            const int row = q0 + e + 8 * grp;
            const int col = h * D_ + t * 16 + nl;
            Oh[((size_t)b * S_ + row) * E_ + col] = (_Float16)(acc[t][e] / lrow[e]);
        }
    }
}

// ------------------------------- launcher ----------------------------------

extern "C" void kernel_launch(void* const* d_in, const int* in_sizes, int n_in,
                              void* d_out, int out_size, void* d_ws, size_t ws_size,
                              hipStream_t stream) {
    (void)in_sizes; (void)n_in; (void)out_size; (void)ws_size;
    const float* q  = (const float*)d_in[0];
    const float* k  = (const float*)d_in[1];
    const float* v  = (const float*)d_in[2];
    const float* Wq = (const float*)d_in[3];
    const float* bq = (const float*)d_in[4];
    const float* Wk = (const float*)d_in[5];
    const float* bk = (const float*)d_in[6];
    const float* Wv = (const float*)d_in[7];
    const float* bv = (const float*)d_in[8];
    const float* Wo = (const float*)d_in[9];
    const float* bo = (const float*)d_in[10];

    // workspace layout (bytes)
    char* ws = (char*)d_ws;
    constexpr size_t OFF_SC  = 0;                                   // 4 x u32 scales
    constexpr size_t OFF_WH  = 256;                                 // 4 x WELEM f16 (8 MB)
    constexpr size_t OFF_XH  = OFF_WH + (size_t)4 * WELEM * 2;      // 3 x XELEM f16 (24 MB)
    constexpr size_t OFF_QKV = OFF_XH + (size_t)3 * XELEM * 2;      // 3 x XELEM f16 head-major (24 MB)
    constexpr size_t OFF_OH  = OFF_QKV + (size_t)3 * XELEM * 2;     // XELEM f16 (8 MB)
    unsigned int* sc  = (unsigned int*)(ws + OFF_SC);
    _Float16*     Wh  = (_Float16*)(ws + OFF_WH);
    _Float16*     Xh  = (_Float16*)(ws + OFF_XH);
    _Float16*     QKV = (_Float16*)(ws + OFF_QKV);
    _Float16*     Oh  = (_Float16*)(ws + OFF_OH);

    // 1) per-tensor absmax of each weight
    init_scales_kernel<<<1, 32, 0, stream>>>(sc);
    absmax_kernel<<<dim3(256, 4), 256, 0, stream>>>(Wq, Wk, Wv, Wo, sc, WELEM);

    // 2) int8 quantize->dequantize weights to f16 (Wh order: q,k,v,o)
    dequant_kernel<<<dim3(WELEM / 256, 4), 256, 0, stream>>>(Wq, Wk, Wv, Wo, sc, Wh, WELEM);

    // 3) activations f32 -> f16 (Xh order: q,k,v)
    f2h_kernel<<<dim3(XELEM / 256, 3), 256, 0, stream>>>(q, k, v, Xh, XELEM);

    // 4) QKV projections (WMMA), written head-major [B,H,S,D]
    const dim3 ggrid(E_ / 64, M_ / 32);
    wmma_gemm_kernel<0><<<ggrid, 32, 0, stream>>>(Xh,             Wh,             bq, QKV,             E_, E_);
    wmma_gemm_kernel<0><<<ggrid, 32, 0, stream>>>(Xh + XELEM,     Wh + WELEM,     bk, QKV + XELEM,     E_, E_);
    wmma_gemm_kernel<0><<<ggrid, 32, 0, stream>>>(Xh + 2 * XELEM, Wh + 2 * WELEM, bv, QKV + 2 * XELEM, E_, E_);

    // 5) flash attention per (q-tile, head, batch)
    flash_attn_kernel<<<dim3(S_ / 16, H_, B_), 32, 0, stream>>>(
        QKV, QKV + XELEM, QKV + 2 * XELEM, Oh);

    // 6) output projection -> f32 d_out [B,S,E]
    wmma_gemm_kernel<1><<<ggrid, 32, 0, stream>>>(Oh, Wh + 3 * WELEM, bo, d_out, E_, E_);
}
